// RNN_2164663517863
// MI455X (gfx1250) — matrix operand order repaired and checked by reading, not compiled, not run on hardware
//
#include <hip/hip_runtime.h>
#include <hip/hip_bf16.h>
#include <math.h>
#include <stdint.h>

// Problem constants (from the reference): T=256 steps, B=256 batch, IN=H=1024.
#define T_STEPS 256
#define BATCH   256
#define IN_DIM  1024
#define H_DIM   1024

// LDS staging for the big GEMM: K-step of 32 bf16 (64B per row), row stride
// padded to 80B -> 16B-aligned chunks and conflict-free ds_load_b128 reads
// (bank(m) = m*20 mod 64 is a permutation over the 16 fragment lanes).
#define KSTEP       32
#define LDS_STRIDE  80
#define STAGE_BYTES (128 * LDS_STRIDE)   // 10 KB per matrix per stage
#define SMEM_BYTES  (4 * STAGE_BYTES)    // A0,B0,A1,B1 = 40 KB

typedef __attribute__((ext_vector_type(16))) __bf16 v16bf;
typedef __attribute__((ext_vector_type(8)))  __bf16 v8bf;
typedef __attribute__((ext_vector_type(4)))  __bf16 v4bf;
typedef __attribute__((ext_vector_type(8)))  float  v8f;

typedef __attribute__((address_space(3))) const v8bf*  as3_v8bf;
typedef __attribute__((address_space(3))) const v16bf* as3_v16bf;

// ---------------------------------------------------------------------------
// tanh via the gfx1250 transcendental unit when available.
// ---------------------------------------------------------------------------
__device__ __forceinline__ float fast_tanh(float x) {
#if __has_builtin(__builtin_amdgcn_tanhf)
  return __builtin_amdgcn_tanhf(x);
#else
  return tanhf(x);
#endif
}

// ---------------------------------------------------------------------------
// fp32 -> bf16 conversion, 4 elements per thread iteration (float4 loads).
// ---------------------------------------------------------------------------
__global__ void __launch_bounds__(256)
rnn_cvt_f32_to_bf16(const float4* __restrict__ in, v4bf* __restrict__ out, long n4) {
  long i = (long)blockIdx.x * blockDim.x + threadIdx.x;
  long stride = (long)gridDim.x * blockDim.x;
  for (; i < n4; i += stride) {
    float4 f = in[i];
    v4bf o;
    o[0] = (__bf16)f.x;
    o[1] = (__bf16)f.y;
    o[2] = (__bf16)f.z;
    o[3] = (__bf16)f.w;
    out[i] = o;
  }
}

// ---------------------------------------------------------------------------
// CDNA5 async copy: 16 bytes/lane, global -> LDS, tracked by ASYNCcnt.
// GV form: vdst = 32-bit LDS byte offset, vaddr = 64-bit global address.
// ---------------------------------------------------------------------------
__device__ __forceinline__ void async_copy_b128(unsigned lds_off, const void* gaddr) {
  asm volatile("global_load_async_to_lds_b128 %0, %1, off"
               :: "v"(lds_off), "v"(gaddr)
               : "memory");
}

// Issue one double-buffer stage: 128 rows x 64B of A and of B (4 async ops
// per thread, all 256 threads cooperating -> ASYNCcnt += 4 per wave).
__device__ __forceinline__ void issue_stage(const __bf16* __restrict__ A,
                                            const __bf16* __restrict__ W,
                                            int K, int mTile, int nTile, int kb,
                                            unsigned sA, unsigned sB, int tid) {
#pragma unroll
  for (int rep = 0; rep < 2; ++rep) {
    const int chunk = rep * 256 + tid;   // 0..511
    const int row = chunk >> 2;          // 0..127
    const int c16 = (chunk & 3) * 16;    // 16B chunk within the 64B K-slab
    async_copy_b128(sA + row * LDS_STRIDE + c16,
                    (const char*)(A + (size_t)(mTile + row) * K + kb) + c16);
    async_copy_b128(sB + row * LDS_STRIDE + c16,
                    (const char*)(W + (size_t)(nTile + row) * K + kb) + c16);
  }
}

// ---------------------------------------------------------------------------
// Fragment loaders for the CDNA5 WMMA operands, A and B in the same contiguous k order (CDNA5 ISA).
// A (16x32, MxK): lane {lh=lane>>4, lm=lane&15}: K chunks at lh*16 and lh*16+8 (16 contiguous K).
// B (32x16, KxN): with W row-major [N][K], lane reads 16 contiguous K.
// ---------------------------------------------------------------------------
__device__ __forceinline__ v16bf load_a_frag(const __bf16* __restrict__ A,
                                             int K, int m, int kb, int lh) {
  const __bf16* row = A + (size_t)m * K + kb;
  v8bf lo = *(const v8bf*)(row + lh * 16);
  v8bf hi = *(const v8bf*)(row + lh * 16 + 8);
  return __builtin_shufflevector(lo, hi, 0, 1, 2, 3, 4, 5, 6, 7,
                                         8, 9, 10, 11, 12, 13, 14, 15);
}

__device__ __forceinline__ v16bf load_b_frag(const __bf16* __restrict__ W,
                                             int K, int n, int kb, int lh) {
  return *(const v16bf*)(W + (size_t)n * K + kb + lh * 16);
}

// Same fragments, sourced from the LDS stage (lower 32 bits of a generic
// shared-memory address are the wave-relative LDS byte offset on CDNA5).
__device__ __forceinline__ v16bf lds_a_frag(unsigned sA, int m, int lh) {
  const unsigned off = sA + (unsigned)m * LDS_STRIDE + lh * 32;
  v8bf lo = *(as3_v8bf)(off);
  v8bf hi = *(as3_v8bf)(off + 16);
  return __builtin_shufflevector(lo, hi, 0, 1, 2, 3, 4, 5, 6, 7,
                                         8, 9, 10, 11, 12, 13, 14, 15);
}

__device__ __forceinline__ v16bf lds_b_frag(unsigned sB, int n, int lh) {
  return *(as3_v16bf)(sB + (unsigned)n * LDS_STRIDE + lh * 32);
}

// ---------------------------------------------------------------------------
// Kernel 2: xi = x @ Wi^T + Wi_b, written directly into d_out.
// Block = 8 waves (2 Mrows x 4 Ncols), tile 128x128, LDS double-buffered via
// global_load_async_to_lds_b128 / s_wait_asynccnt. Wave = 64x32 (4x2 frags).
// ---------------------------------------------------------------------------
__global__ void __launch_bounds__(256)
rnn_gemm_xi_kernel(const __bf16* __restrict__ xbf, const __bf16* __restrict__ wibf,
                   const float* __restrict__ bias, float* __restrict__ out,
                   int M, int K, int N) {
  __shared__ __align__(16) char smem[SMEM_BYTES];
  const int tid  = threadIdx.x;
  const int lane = tid & 31;
  const int w    = tid >> 5;
  const int lm   = lane & 15;
  const int lh   = lane >> 4;
  const int mTile = blockIdx.x * 128;
  const int nTile = blockIdx.y * 128;
  const int wm = (w >> 2) * 64;   // wave M offset within tile
  const int wn = (w & 3) * 32;    // wave N offset within tile

  const unsigned lds0 = (unsigned)(uintptr_t)smem;
  const unsigned A0 = lds0;
  const unsigned B0 = lds0 + STAGE_BYTES;
  const unsigned A1 = lds0 + 2 * STAGE_BYTES;
  const unsigned B1 = lds0 + 3 * STAGE_BYTES;

  v8f acc[4][2];
  v8f zero = {};
#pragma unroll
  for (int i = 0; i < 4; ++i)
#pragma unroll
    for (int j = 0; j < 2; ++j) acc[i][j] = zero;

  issue_stage(xbf, wibf, K, mTile, nTile, 0, A0, B0, tid);

  for (int kb = 0; kb < K; kb += KSTEP) {
    const int odd = (kb / KSTEP) & 1;
    const unsigned sA = odd ? A1 : A0;
    const unsigned sB = odd ? B1 : B0;
    if (kb + KSTEP < K) {
      issue_stage(xbf, wibf, K, mTile, nTile, kb + KSTEP,
                  odd ? A0 : A1, odd ? B0 : B1, tid);
      // 4 newest copies outstanding; current stage's 4 are complete.
      asm volatile("s_wait_asynccnt 0x4" ::: "memory");
    } else {
      asm volatile("s_wait_asynccnt 0x0" ::: "memory");
    }
    __syncthreads();

    v16bf a[4];
    v16bf b[2];
#pragma unroll
    for (int i = 0; i < 4; ++i) a[i] = lds_a_frag(sA, wm + i * 16 + lm, lh);
#pragma unroll
    for (int j = 0; j < 2; ++j) b[j] = lds_b_frag(sB, wn + j * 16 + lm, lh);
#pragma unroll
    for (int i = 0; i < 4; ++i) {
#pragma unroll
      for (int j = 0; j < 2; ++j) {
        acc[i][j] = __builtin_amdgcn_wmma_f32_16x16x32_bf16(
            false, a[i], false, b[j], (short)0, acc[i][j], false, false);
      }
    }
    __syncthreads();  // all reads of this stage done before it is re-filled
  }

#pragma unroll
  for (int i = 0; i < 4; ++i) {
#pragma unroll
    for (int j = 0; j < 2; ++j) {
      const int n = nTile + wn + j * 16 + lm;
      const float bn = bias[n];
#pragma unroll
      for (int r = 0; r < 8; ++r) {
        const int m = mTile + wm + i * 16 + lh * 8 + r;
        out[(size_t)m * N + n] = acc[i][j][r] + bn;
      }
    }
  }
}

// ---------------------------------------------------------------------------
// Kernel 3 (launched T times): h = tanh(xi_t + h_prev @ Wh^T + b).
// The scan is the serial critical path, so maximize per-step parallelism:
// wave tile 16x32 (1 A frag, 2 B frags), block 64x64, grid 4x16 = 512 waves.
// h (0.5 MB bf16) and Wh (2 MB bf16) are L2-resident -> direct global loads.
// Reads xi_t from d_out[t] and overwrites it with h_t; bf16 copy for t+1.
// ---------------------------------------------------------------------------
__global__ void __launch_bounds__(256)
rnn_step_kernel(const __bf16* __restrict__ hprev, const __bf16* __restrict__ whbf,
                const float* __restrict__ bias, float* __restrict__ io,
                __bf16* __restrict__ hnext, int M, int K, int N) {
  const int lane = threadIdx.x & 31;
  const int w    = threadIdx.x >> 5;
  const int lm   = lane & 15;
  const int lh   = lane >> 4;
  const int m0 = blockIdx.x * 64 + (w & 3) * 16;
  const int n0 = blockIdx.y * 64 + (w >> 2) * 32;

  v8f acc[2];
  v8f zero = {};
  acc[0] = zero;
  acc[1] = zero;

  for (int kb = 0; kb < K; kb += 32) {
    if (kb + 32 < K) {
      __builtin_prefetch(hprev + (size_t)(m0 + lm) * K + kb + 32, 0, 3);
      __builtin_prefetch(whbf + (size_t)(n0 + lm) * K + kb + 32, 0, 3);
    }
    v16bf a  = load_a_frag(hprev, K, m0 + lm, kb, lh);
    v16bf b0 = load_b_frag(whbf, K, n0 + lm, kb, lh);
    v16bf b1 = load_b_frag(whbf, K, n0 + 16 + lm, kb, lh);
    acc[0] = __builtin_amdgcn_wmma_f32_16x16x32_bf16(
        false, a, false, b0, (short)0, acc[0], false, false);
    acc[1] = __builtin_amdgcn_wmma_f32_16x16x32_bf16(
        false, a, false, b1, (short)0, acc[1], false, false);
  }

#pragma unroll
  for (int j = 0; j < 2; ++j) {
    const int n = n0 + j * 16 + lm;
    const float bn = bias[n];
#pragma unroll
    for (int r = 0; r < 8; ++r) {
      const int m = m0 + lh * 8 + r;
      const size_t idx = (size_t)m * N + n;
      float v = fast_tanh(acc[j][r] + io[idx] + bn);
      io[idx] = v;                 // hiddens[t] output (fp32)
      hnext[idx] = (__bf16)v;      // A-matrix for step t+1
    }
  }
}

// ---------------------------------------------------------------------------
// Host-side launch. Inputs (setup_inputs order):
//   d_in[0]=x [T,B,IN] f32, d_in[1]=h [B,H] f32, d_in[2]=Wi_w [H,IN] f32,
//   d_in[3]=Wi_b [H] f32,   d_in[4]=Wh_w [H,H] f32, d_in[5]=Wh_b [H] f32.
// d_out = hiddens [T,B,H] f32. Workspace: bf16 x / Wi / Wh + h ping-pong.
// Every buffer is rewritten from inputs on every call -> graph-safe.
// ---------------------------------------------------------------------------
extern "C" void kernel_launch(void* const* d_in, const int* in_sizes, int n_in,
                              void* d_out, int out_size, void* d_ws, size_t ws_size,
                              hipStream_t stream) {
  const float* x    = (const float*)d_in[0];
  const float* h0   = (const float*)d_in[1];
  const float* Wi_w = (const float*)d_in[2];
  const float* Wi_b = (const float*)d_in[3];
  const float* Wh_w = (const float*)d_in[4];
  const float* Wh_b = (const float*)d_in[5];
  float* out = (float*)d_out;

  char* p = (char*)d_ws;
  __bf16* xbf   = (__bf16*)p; p += (size_t)T_STEPS * BATCH * IN_DIM * sizeof(__bf16);
  __bf16* wibf  = (__bf16*)p; p += (size_t)H_DIM * IN_DIM * sizeof(__bf16);
  __bf16* whbf  = (__bf16*)p; p += (size_t)H_DIM * H_DIM * sizeof(__bf16);
  __bf16* hping = (__bf16*)p; p += (size_t)BATCH * H_DIM * sizeof(__bf16);
  __bf16* hpong = (__bf16*)p;

  // Phase 1: fp32 -> bf16 conversions.
  const long nx = (long)T_STEPS * BATCH * IN_DIM;
  rnn_cvt_f32_to_bf16<<<4096, 256, 0, stream>>>((const float4*)x, (v4bf*)xbf, nx / 4);
  rnn_cvt_f32_to_bf16<<<512, 256, 0, stream>>>((const float4*)Wi_w, (v4bf*)wibf,
                                               (long)H_DIM * IN_DIM / 4);
  rnn_cvt_f32_to_bf16<<<512, 256, 0, stream>>>((const float4*)Wh_w, (v4bf*)whbf,
                                               (long)H_DIM * H_DIM / 4);
  rnn_cvt_f32_to_bf16<<<64, 256, 0, stream>>>((const float4*)h0, (v4bf*)hping,
                                              (long)BATCH * H_DIM / 4);

  // Phase 2: xi = x @ Wi^T + bi for all timesteps, written into d_out.
  rnn_gemm_xi_kernel<<<dim3((T_STEPS * BATCH) / 128, H_DIM / 128), 256, 0, stream>>>(
      xbf, wibf, Wi_b, out, T_STEPS * BATCH, IN_DIM, H_DIM);

  // Phase 3: sequential scan; each step is one kernel on the stream.
  for (int t = 0; t < T_STEPS; ++t) {
    const __bf16* hin = (t & 1) ? hpong : hping;
    __bf16* hout      = (t & 1) ? hping : hpong;
    rnn_step_kernel<<<dim3(BATCH / 64, H_DIM / 64), 256, 0, stream>>>(
        hin, whbf, Wh_b, out + (size_t)t * BATCH * H_DIM, hout,
        BATCH, H_DIM, H_DIM);
  }
}
